// Taco2ARNet_54013508715185
// MI455X (gfx1250) — matrix operand order repaired and checked by reading, not compiled, not run on hardware
//
#include <hip/hip_runtime.h>
#include <hip/hip_bf16.h>

typedef __bf16 bf16_t;
typedef bf16_t v16bf __attribute__((ext_vector_type(16)));
typedef bf16_t v8bf  __attribute__((ext_vector_type(8)));
typedef float  v8f   __attribute__((ext_vector_type(8)));

#define NWG    64
#define TPB    256
#define NWAVES (NWG * 8)

// problem dims
#define Bz 32
#define Tz 1000
#define Dz 1024
#define Hz 1024
#define Az 256
#define Oz 80
#define K0 2304   // D + A + H  (u0 = [x_t | prenet | za])
#define K1 2048   // H + H      (u1 = [za | zb])
#define G4 4096   // 4*H gates
#define KSPL 4    // K-split for gate GEMMs (one slice per WG, 4-way)
#define KC0 (K0 / KSPL)   // 576
#define KC1 (K1 / KSPL)   // 512
#define LPAD 8            // LDS row pad (elements) to spread banks

// ---- workspace layout (bytes, every block multiple of 256) ----
constexpr size_t OFF_BAR  = 0;
constexpr size_t OFF_W0   = 256;
constexpr size_t OFF_W1   = OFF_W0   + (size_t)4096 * 2304 * 2;   // W0cat bf16
constexpr size_t OFF_WP0  = OFF_W1   + (size_t)4096 * 2048 * 2;   // W1cat bf16
constexpr size_t OFF_WP1  = OFF_WP0  + (size_t)1024 * 1024 * 2;
constexpr size_t OFF_W1P  = OFF_WP1  + (size_t)1024 * 1024 * 2;   // prenet1 padded 256x96
constexpr size_t OFF_W2P  = OFF_W1P  + (size_t)256 * 96 * 2;      // prenet2 256x256
constexpr size_t OFF_WPJ  = OFF_W2P  + (size_t)256 * 256 * 2;     // wproj 80x1024
constexpr size_t OFF_U0   = OFF_WPJ  + (size_t)80 * 1024 * 2;     // 32x2304 bf16
constexpr size_t OFF_U1   = OFF_U0   + (size_t)32 * 2304 * 2;     // 32x2048 bf16
constexpr size_t OFF_PREV = OFF_U1   + (size_t)32 * 2048 * 2;     // 32x96  bf16 (padded AR input)
constexpr size_t OFF_G0P  = OFF_PREV + (size_t)32 * 96 * 2;       // 4 x 32x4096 f32 partials
constexpr size_t OFF_G1P  = OFF_G0P  + (size_t)KSPL * 32 * 4096 * 4;
constexpr size_t OFF_C0   = OFF_G1P  + (size_t)KSPL * 32 * 4096 * 4;
constexpr size_t OFF_C1   = OFF_C0   + (size_t)32 * 1024 * 4;
constexpr size_t OFF_H0   = OFF_C1   + (size_t)32 * 1024 * 4;     // 32x1024 bf16
constexpr size_t OFF_H1   = OFF_H0   + (size_t)32 * 1024 * 2;
constexpr size_t WS_NEED  = OFF_H1   + (size_t)32 * 1024 * 2;     // ~45 MB

// ---------------- fragment loads (ISA 7.12.2 layouts) ----------------
// A 16x32 bf16: lanes 0-15 row M=lr hold K {k..k+7, k+16..k+23};
//               lanes 16-31 row M=lr hold K {k+8..k+15, k+24..k+31}.
__device__ __forceinline__ v16bf load_fragA(const bf16_t* p) {
  union { v16bf v; v8bf h[2]; } u;
  u.h[0] = *reinterpret_cast<const v8bf*>(p);        // 8 bf16, 16B
  u.h[1] = *reinterpret_cast<const v8bf*>(p + 16);   // +16 elements
  return u.v;
}
// B 32x16 bf16: lanes 0-15 col N=lr hold K {k..k+7, k+16..k+23}; lanes 16-31 K {k+8..k+15, k+24..k+31}.
__device__ __forceinline__ v16bf load_fragB(const bf16_t* p) {
  union { v16bf v; v8bf h[2]; } u;
  u.h[0] = *reinterpret_cast<const v8bf*>(p);
  u.h[1] = *reinterpret_cast<const v8bf*>(p + 16);
  return u.v;
}

// 32x16 output tile: acc0 rows 0..15, acc1 rows 16..31. W row-major (N x K) => B col = W row.
// Separate k-origins for A (k0a, e.g. 0 inside an LDS slice) and B (k0b inside the full W row).
__device__ __forceinline__ void mm_tile(const bf16_t* __restrict__ A, int lda, int k0a,
                                        const bf16_t* __restrict__ W, int ldw, int k0b,
                                        int n0, int ksteps, v8f& acc0, v8f& acc1) {
  const int lane = threadIdx.x & 31;
  const int half = lane >> 4;
  const int lr   = lane & 15;
  const bf16_t* a0p = A + (size_t)lr * lda + half * 8 + k0a;
  const bf16_t* a1p = a0p + (size_t)16 * lda;
  const bf16_t* bp  = W + (size_t)(n0 + lr) * ldw + half * 8 + k0b;
  for (int i = 0; i < ksteps; ++i) {
    v16bf a0 = load_fragA(a0p + i * 32);
    v16bf a1 = load_fragA(a1p + i * 32);
    v16bf b  = load_fragB(bp  + i * 32);
    acc0 = __builtin_amdgcn_wmma_f32_16x16x32_bf16(false, a0, false, b, (short)0, acc0, false, false);
    acc1 = __builtin_amdgcn_wmma_f32_16x16x32_bf16(false, a1, false, b, (short)0, acc1, false, false);
  }
}

__device__ __forceinline__ void store_tile_f32(float* __restrict__ dst, int ldo, int n0,
                                               const v8f& a0, const v8f& a1) {
  const int lane = threadIdx.x & 31;
  const int half = lane >> 4;
  const int lr   = lane & 15;
#pragma unroll
  for (int r = 0; r < 8; ++r) {
    int m = r + 8 * half;                 // C/D layout: VGPR r -> M=r (lanes<16) / M=r+8
    dst[(size_t)m        * ldo + n0 + lr] = a0[r];
    dst[(size_t)(16 + m) * ldo + n0 + lr] = a1[r];
  }
}

// ---------------- device-wide barrier (all 64 WGs co-resident) ----------------
__device__ __forceinline__ void grid_sync(unsigned* cnt, unsigned* gen) {
  __syncthreads();
  if (threadIdx.x == 0) {
    __threadfence();
    unsigned g = __hip_atomic_load(gen, __ATOMIC_RELAXED, __HIP_MEMORY_SCOPE_AGENT);
    unsigned a = __hip_atomic_fetch_add(cnt, 1u, __ATOMIC_ACQ_REL, __HIP_MEMORY_SCOPE_AGENT);
    if (a == NWG - 1) {
      __hip_atomic_store(cnt, 0u, __ATOMIC_RELAXED, __HIP_MEMORY_SCOPE_AGENT);
      __hip_atomic_fetch_add(gen, 1u, __ATOMIC_RELEASE, __HIP_MEMORY_SCOPE_AGENT);
    } else {
      while (__hip_atomic_load(gen, __ATOMIC_ACQUIRE, __HIP_MEMORY_SCOPE_AGENT) == g) {
        __builtin_amdgcn_s_sleep(2);
      }
    }
  }
  __syncthreads();
}

__device__ __forceinline__ float sigm(float x) { return 1.f / (1.f + __expf(-x)); }

// sum KSPL K-split partials + biases, LSTM cell elementwise, emit h as bf16
__device__ __forceinline__ void cell_stage(const float* __restrict__ gp,
                                           const float* __restrict__ bih,
                                           const float* __restrict__ bhh,
                                           float* __restrict__ c, bf16_t* __restrict__ hb, int tid) {
  for (int e = tid; e < Bz * Hz; e += NWG * TPB) {
    int m = e >> 10, j = e & 1023;
    float gi = bih[j]        + bhh[j];
    float gf = bih[j + 1024] + bhh[j + 1024];
    float gg = bih[j + 2048] + bhh[j + 2048];
    float go = bih[j + 3072] + bhh[j + 3072];
#pragma unroll
    for (int p = 0; p < KSPL; ++p) {
      const float* g = gp + (size_t)p * Bz * G4 + (size_t)m * G4;
      gi += g[j]; gf += g[j + 1024]; gg += g[j + 2048]; go += g[j + 3072];
    }
    float cn = sigm(gf) * c[e] + sigm(gi) * tanhf(gg);
    c[e] = cn;
    hb[e] = (bf16_t)(sigm(go) * tanhf(cn));
  }
}

// async copy one K-slice of U (32 x Kc, row stride ldu, col origin k0) into LDS (row stride Kc+LPAD)
__device__ __forceinline__ void async_stage_A(const bf16_t* __restrict__ U, int ldu, int k0,
                                              int Kc, bf16_t* sA) {
  const int nch = Kc >> 3;                 // 16B chunks per row
  for (int c = threadIdx.x; c < 32 * nch; c += TPB) {
    int m = c / nch, q = c - m * nch;
    unsigned lo = (unsigned)(size_t)&sA[(size_t)m * (Kc + LPAD) + q * 8];
    unsigned long long ga =
        (unsigned long long)(size_t)(U + (size_t)m * ldu + k0 + q * 8);
    asm volatile("global_load_async_to_lds_b128 %0, %1, off"
                 :: "v"(lo), "v"(ga) : "memory");
  }
  asm volatile("s_wait_asynccnt 0x0" ::: "memory");   // my async loads into LDS done
  __syncthreads();                                    // everyone's done
}

// ---------------- prologue: pack all weights f32 -> bf16, K-concatenated ----------------
__global__ void __launch_bounds__(256) pack_weights(
    const float* __restrict__ wih0, const float* __restrict__ whh0,
    const float* __restrict__ wih1, const float* __restrict__ whh1,
    const float* __restrict__ wp0,  const float* __restrict__ wp1,
    const float* __restrict__ pw1,  const float* __restrict__ pw2,
    const float* __restrict__ wproj, char* __restrict__ ws) {
  bf16_t* W0  = (bf16_t*)(ws + OFF_W0);
  bf16_t* W1  = (bf16_t*)(ws + OFF_W1);
  bf16_t* P0  = (bf16_t*)(ws + OFF_WP0);
  bf16_t* P1  = (bf16_t*)(ws + OFF_WP1);
  bf16_t* W1P = (bf16_t*)(ws + OFF_W1P);
  bf16_t* W2P = (bf16_t*)(ws + OFF_W2P);
  bf16_t* WPJ = (bf16_t*)(ws + OFF_WPJ);
  long stride = (long)gridDim.x * blockDim.x;
  long t0 = (long)blockIdx.x * blockDim.x + threadIdx.x;
  for (long i = t0; i < 4096L * 2304; i += stride) {      // W0cat = [wih0 | whh0]
    long n = i / 2304, k = i - n * 2304;
    W0[i] = (bf16_t)((k < 1280) ? wih0[n * 1280 + k] : whh0[n * 1024 + (k - 1280)]);
  }
  for (long i = t0; i < 4096L * 2048; i += stride) {      // W1cat = [wih1 | whh1]
    long n = i >> 11, k = i & 2047;
    W1[i] = (bf16_t)((k < 1024) ? wih1[(n << 10) + k] : whh1[(n << 10) + (k - 1024)]);
  }
  for (long i = t0; i < 1024L * 1024; i += stride) { P0[i] = (bf16_t)wp0[i]; P1[i] = (bf16_t)wp1[i]; }
  for (long i = t0; i < 256L * 96; i += stride) {         // prenet1 K padded 80 -> 96
    long n = i / 96, k = i - n * 96;
    W1P[i] = (bf16_t)((k < 80) ? pw1[n * 80 + k] : 0.f);
  }
  for (long i = t0; i < 256L * 256; i += stride) W2P[i] = (bf16_t)pw2[i];
  for (long i = t0; i < 80L * 1024; i += stride) WPJ[i] = (bf16_t)wproj[i];
}

// ---------------- persistent AR decoder ----------------
__global__ void __launch_bounds__(TPB) taco2_persistent(
    const float* __restrict__ cond,
    const float* __restrict__ pb1,  const float* __restrict__ pb2,
    const float* __restrict__ bih0, const float* __restrict__ bhh0, const float* __restrict__ bp0,
    const float* __restrict__ bih1, const float* __restrict__ bhh1, const float* __restrict__ bp1,
    const float* __restrict__ bproj, const float* __restrict__ mean, const float* __restrict__ scale,
    float* __restrict__ out, char* __restrict__ ws) {
  // LDS: gate stages stage the A-operand K-slice here (max 32 x (576+8) bf16 = 36.5 KB);
  // prenet reuses it for the a1 intermediate (32 x 264 bf16).
  __shared__ __align__(16) bf16_t sA[32 * (KC0 + LPAD)];

  unsigned* cnt = (unsigned*)(ws + OFF_BAR);
  unsigned* gen = (unsigned*)(ws + OFF_BAR + 4);
  const bf16_t* W0  = (const bf16_t*)(ws + OFF_W0);
  const bf16_t* W1  = (const bf16_t*)(ws + OFF_W1);
  const bf16_t* P0  = (const bf16_t*)(ws + OFF_WP0);
  const bf16_t* P1  = (const bf16_t*)(ws + OFF_WP1);
  const bf16_t* W1P = (const bf16_t*)(ws + OFF_W1P);
  const bf16_t* W2P = (const bf16_t*)(ws + OFF_W2P);
  const bf16_t* WPJ = (const bf16_t*)(ws + OFF_WPJ);
  bf16_t* u0    = (bf16_t*)(ws + OFF_U0);
  bf16_t* u1    = (bf16_t*)(ws + OFF_U1);
  bf16_t* prevn = (bf16_t*)(ws + OFF_PREV);
  float*  g0p   = (float*)(ws + OFF_G0P);
  float*  g1p   = (float*)(ws + OFF_G1P);
  float*  c0    = (float*)(ws + OFF_C0);
  float*  c1    = (float*)(ws + OFF_C1);
  bf16_t* h0b   = (bf16_t*)(ws + OFF_H0);
  bf16_t* h1b   = (bf16_t*)(ws + OFF_H1);

  const int tid  = blockIdx.x * TPB + threadIdx.x;  // 0..16383
  const int gw   = tid >> 5;                        // global wave 0..511
  const int lane = threadIdx.x & 31;
  const int half = lane >> 4;
  const int lr   = lane & 15;
  const int ks   = blockIdx.x & (KSPL - 1);         // this WG's gate K-slice
  const int wvk  = (blockIdx.x >> 2) * 8 + (threadIdx.x >> 5); // wave id within K-slice group: 0..127

  // ---- t=0 state init: c=0, za=0 (u0 tail), u1=0, prevn=0, cond[t=0] -> u0[:,0:1024] ----
  for (int i = tid; i < Bz * Hz; i += NWG * TPB) { c0[i] = 0.f; c1[i] = 0.f; }
  for (int i = tid; i < Bz * Hz; i += NWG * TPB) u0[(size_t)(i >> 10) * K0 + 1280 + (i & 1023)] = (bf16_t)0.f;
  for (int i = tid; i < Bz * K1; i += NWG * TPB) u1[i] = (bf16_t)0.f;
  for (int i = tid; i < Bz * 96; i += NWG * TPB) prevn[i] = (bf16_t)0.f;
  for (int i = tid; i < Bz * Dz; i += NWG * TPB) {
    int m = i >> 10, d = i & 1023;
    u0[(size_t)m * K0 + d] = (bf16_t)cond[(size_t)m * Tz * Dz + d];
  }
  grid_sync(cnt, gen);

  for (int t = 0; t < Tz; ++t) {
    // ---- S12: prenet FC1+ReLU -> sA(LDS), __syncthreads, FC2+ReLU -> u0[:,1024:1280].
    //      Runs entirely in WG 0 (16 tiles per layer, 2 per wave). ----
    if (blockIdx.x == 0) {
      const int w = threadIdx.x >> 5;
#pragma unroll
      for (int i = 0; i < 2; ++i) {
        int n0 = (w * 2 + i) * 16;
        v8f acc0 = {}, acc1 = {};
        mm_tile(prevn, 96, 0, W1P, 96, 0, n0, 3, acc0, acc1);
        float b = pb1[n0 + lr];
#pragma unroll
        for (int r = 0; r < 8; ++r) {
          int m = r + 8 * half;
          float v0 = acc0[r] + b; v0 = v0 > 0.f ? v0 : 0.f;
          float v1 = acc1[r] + b; v1 = v1 > 0.f ? v1 : 0.f;
          sA[(size_t)m * (Az + LPAD) + n0 + lr]        = (bf16_t)v0;
          sA[(size_t)(16 + m) * (Az + LPAD) + n0 + lr] = (bf16_t)v1;
        }
      }
      __syncthreads();
#pragma unroll
      for (int i = 0; i < 2; ++i) {
        int n0 = (w * 2 + i) * 16;
        v8f acc0 = {}, acc1 = {};
        mm_tile(sA, Az + LPAD, 0, W2P, Az, 0, n0, Az / 32, acc0, acc1);
        float b = pb2[n0 + lr];
#pragma unroll
        for (int r = 0; r < 8; ++r) {
          int m = r + 8 * half;
          float v0 = acc0[r] + b; v0 = v0 > 0.f ? v0 : 0.f;
          float v1 = acc1[r] + b; v1 = v1 > 0.f ? v1 : 0.f;
          u0[(size_t)m * K0 + 1024 + n0 + lr]        = (bf16_t)v0;
          u0[(size_t)(16 + m) * K0 + 1024 + n0 + lr] = (bf16_t)v1;
        }
      }
    }
    grid_sync(cnt, gen);

    // ---- S3: gates0 = u0 (32x2304) @ W0cat^T (4096x2304), K-split 4.
    //      A K-slice async-staged to LDS; B streams from L2. 2 N-tiles per wave. ----
    async_stage_A(u0, K0, ks * KC0, KC0, sA);
#pragma unroll
    for (int i = 0; i < 2; ++i) {
      int nt = wvk * 2 + i;                 // 0..255
      v8f acc0 = {}, acc1 = {};
      mm_tile(sA, KC0 + LPAD, 0, W0, K0, ks * KC0, nt * 16, KC0 / 32, acc0, acc1);
      store_tile_f32(g0p + (size_t)ks * Bz * G4, G4, nt * 16, acc0, acc1);
    }
    grid_sync(cnt, gen);

    // ---- S4: LSTM cell 0 elementwise -> c0, h0b ----
    cell_stage(g0p, bih0, bhh0, c0, h0b, tid);
    grid_sync(cnt, gen);

    // ---- S5: za = tanh(h0 @ wp0^T + bp0) -> u0[:,1280:2304] (next step) and u1[:,0:1024];
    //      idle waves convert cond[t+1] -> u0[:,0:1024] (consumed at next step's S3). ----
    if (gw < 64) {
      v8f acc0 = {}, acc1 = {};
      int n0 = gw * 16;
      mm_tile(h0b, Hz, 0, P0, Hz, 0, n0, 32, acc0, acc1);
      float b = bp0[n0 + lr];
#pragma unroll
      for (int r = 0; r < 8; ++r) {
        int m = r + 8 * half;
        bf16_t z0 = (bf16_t)tanhf(acc0[r] + b);
        bf16_t z1 = (bf16_t)tanhf(acc1[r] + b);
        u0[(size_t)m * K0 + 1280 + n0 + lr]        = z0;
        u0[(size_t)(16 + m) * K0 + 1280 + n0 + lr] = z1;
        u1[(size_t)m * K1 + n0 + lr]               = z0;
        u1[(size_t)(16 + m) * K1 + n0 + lr]        = z1;
      }
    } else if (t + 1 < Tz) {
      for (int i = (gw - 64) * 32 + lane; i < Bz * Dz; i += (NWAVES - 64) * 32) {
        int m = i >> 10, d = i & 1023;
        u0[(size_t)m * K0 + d] = (bf16_t)cond[(size_t)m * Tz * Dz + (size_t)(t + 1) * Dz + d];
      }
    }
    grid_sync(cnt, gen);

    // ---- S6: gates1 = u1 (32x2048) @ W1cat^T (4096x2048), K-split 4, LDS-staged A ----
    async_stage_A(u1, K1, ks * KC1, KC1, sA);
#pragma unroll
    for (int i = 0; i < 2; ++i) {
      int nt = wvk * 2 + i;
      v8f acc0 = {}, acc1 = {};
      mm_tile(sA, KC1 + LPAD, 0, W1, K1, ks * KC1, nt * 16, KC1 / 32, acc0, acc1);
      store_tile_f32(g1p + (size_t)ks * Bz * G4, G4, nt * 16, acc0, acc1);
    }
    grid_sync(cnt, gen);

    // ---- S7: LSTM cell 1 elementwise -> c1, h1b ----
    cell_stage(g1p, bih1, bhh1, c1, h1b, tid);
    grid_sync(cnt, gen);

    // ---- S8a: zb = tanh(h1 @ wp1^T + bp1) -> u1[:,1024:2048] ----
    if (gw < 64) {
      v8f acc0 = {}, acc1 = {};
      int n0 = gw * 16;
      mm_tile(h1b, Hz, 0, P1, Hz, 0, n0, 32, acc0, acc1);
      float b = bp1[n0 + lr];
#pragma unroll
      for (int r = 0; r < 8; ++r) {
        int m = r + 8 * half;
        u1[(size_t)m * K1 + 1024 + n0 + lr]        = (bf16_t)tanhf(acc0[r] + b);
        u1[(size_t)(16 + m) * K1 + 1024 + n0 + lr] = (bf16_t)tanhf(acc1[r] + b);
      }
    }
    grid_sync(cnt, gen);

    // ---- S8b: pred = zb @ wproj^T + bproj -> out[:,t,:]; prevn = (pred-mean)/scale ----
    if (gw < 5) {
      v8f acc0 = {}, acc1 = {};
      int n0 = gw * 16;
      mm_tile(u1 + 1024, K1, 0, WPJ, Hz, 0, n0, 32, acc0, acc1);
      float bj = bproj[n0 + lr], mn = mean[n0 + lr], sc = scale[n0 + lr];
#pragma unroll
      for (int r = 0; r < 8; ++r) {
        int m = r + 8 * half;
        float p0 = acc0[r] + bj;
        float p1 = acc1[r] + bj;
        out[(size_t)m * Tz * Oz + (size_t)t * Oz + n0 + lr]        = p0;
        out[(size_t)(16 + m) * Tz * Oz + (size_t)t * Oz + n0 + lr] = p1;
        prevn[(size_t)m * 96 + n0 + lr]        = (bf16_t)((p0 - mn) / sc);
        prevn[(size_t)(16 + m) * 96 + n0 + lr] = (bf16_t)((p1 - mn) / sc);
      }
    }
    grid_sync(cnt, gen);
  }
}

extern "C" void kernel_launch(void* const* d_in, const int* in_sizes, int n_in,
                              void* d_out, int out_size, void* d_ws, size_t ws_size,
                              hipStream_t stream) {
  (void)in_sizes; (void)n_in; (void)out_size;
  if (ws_size < WS_NEED) return;
  const float* cond  = (const float*)d_in[0];
  const float* pw1   = (const float*)d_in[1];
  const float* pb1   = (const float*)d_in[2];
  const float* pw2   = (const float*)d_in[3];
  const float* pb2   = (const float*)d_in[4];
  const float* wih0  = (const float*)d_in[5];
  const float* whh0  = (const float*)d_in[6];
  const float* bih0  = (const float*)d_in[7];
  const float* bhh0  = (const float*)d_in[8];
  const float* wp0   = (const float*)d_in[9];
  const float* bp0   = (const float*)d_in[10];
  const float* wih1  = (const float*)d_in[11];
  const float* whh1  = (const float*)d_in[12];
  const float* bih1  = (const float*)d_in[13];
  const float* bhh1  = (const float*)d_in[14];
  const float* wp1   = (const float*)d_in[15];
  const float* bp1   = (const float*)d_in[16];
  const float* wproj = (const float*)d_in[17];
  const float* bproj = (const float*)d_in[18];
  const float* mean  = (const float*)d_in[19];
  const float* scale = (const float*)d_in[20];
  char* ws = (char*)d_ws;

  hipMemsetAsync(ws, 0, 256, stream);  // barrier counters
  pack_weights<<<512, 256, 0, stream>>>(wih0, whh0, wih1, whh1, wp0, wp1, pw1, pw2, wproj, ws);
  taco2_persistent<<<NWG, TPB, 0, stream>>>(cond, pb1, pb2, bih0, bhh0, bp0,
                                            bih1, bhh1, bp1, bproj, mean, scale,
                                            (float*)d_out, ws);
}
